// TransformerModel_8237747274215
// MI455X (gfx1250) — matrix operand order repaired and checked by reading, not compiled, not run on hardware
//
#include <hip/hip_runtime.h>
#include <stdint.h>

// ---------------------------------------------------------------------------
// GPT-style transformer forward for MI455X (gfx1250, wave32, WMMA).
// bf16 WMMA GEMMs (fp32 accumulate) with double-buffered async global->LDS
// staging (GLOBAL_LOAD_ASYNC_TO_LDS_B128 + s_wait_asynccnt), pre-transposed
// bf16 weights so both GEMM operands stage as straight async copies,
// flash-attention with online softmax (V staged transposed so P.V fragments
// are contiguous ds_load_b128), template-specialized fused epilogues.
// Workload is HBM-bound (logits write dominates); bf16 halves weight traffic
// and all bf16 weights (~70MB) stay resident in the 192MB L2.
// ---------------------------------------------------------------------------

#define LNUM   6
#define HNUM   8
#define DMODEL 512
#define HSZ    64
#define VOCAB  32000
#define TSEQ   2048
#define BATCH  2
#define ROWS   (BATCH * TSEQ)   // 4096
#define DFF    (4 * DMODEL)     // 2048

typedef __attribute__((ext_vector_type(16))) __bf16 v16bf;
typedef __attribute__((ext_vector_type(8)))  float  v8f;

union Frag { v16bf v; uint32_t u[8]; uint16_t h[16]; };
union U4   { uint4 q; uint16_t h[8]; };

__device__ __forceinline__ uint16_t f2bf(float f) {
  uint32_t x = __float_as_uint(f);
  uint32_t r = (x + 0x7fffu + ((x >> 16) & 1u)) >> 16;  // round-nearest-even
  return (uint16_t)r;
}

__device__ __forceinline__ v8f wmma_bf16(const Frag& a, const Frag& b, v8f c) {
  // D = A(16x32 bf16) * B(32x16 bf16) + C(16x16 f32)
  return __builtin_amdgcn_wmma_f32_16x16x32_bf16(false, a.v, false, b.v,
                                                 (short)0, c, false, false);
}

// Async 16-byte global->LDS copy (CDNA5 GLOBAL_LOAD_ASYNC_TO_LDS_B128,
// tracked by ASYNCcnt). lds pointer's low 32 bits are the LDS byte offset
// (LDS aperture address).
__device__ __forceinline__ void async_copy_b128(uint16_t* lds_dst, const uint16_t* gsrc) {
  uint32_t l = (uint32_t)(uintptr_t)lds_dst;
  uint64_t g = (uint64_t)(uintptr_t)gsrc;
  asm volatile("global_load_async_to_lds_b128 %0, %1, off" :: "v"(l), "v"(g) : "memory");
}
__device__ __forceinline__ void wait_async0() {
  asm volatile("s_wait_asynccnt 0" ::: "memory");
}

// A fragment (16-bit A, 16x32): caller passes pointer to row m (= lane&15).
// Element e -> K = koff + (lane/16)*16 + e; loaded as 8 dwords.
__device__ __forceinline__ void load_a(Frag& f, const uint16_t* row, int koff, int grp) {
#pragma unroll
  for (int i = 0; i < 8; ++i) {
    int k0 = koff + (grp << 4) + (i << 1);
    f.u[i] = *(const uint32_t*)(row + k0);
  }
}

// B fragment from a K-contiguous (transposed) layout: lane = N row, 16
// contiguous halfs at koff + 16*(lane/16) (merged into ds_load_b128 pairs).
__device__ __forceinline__ void load_b_contig(Frag& f, const uint16_t* row, int koff, int grp) {
  const uint32_t* p = (const uint32_t*)(row + koff + (grp << 4));
#pragma unroll
  for (int i = 0; i < 8; ++i) f.u[i] = p[i];
}

// ---------------------------------------------------------------------------
// Embedding: x[b,t,:] = tok_emb[idx[b,t],:] + pos_emb[t,:]
// ---------------------------------------------------------------------------
__global__ void embed_kernel(const int* __restrict__ idx, const float* __restrict__ tok,
                             const float* __restrict__ pos, float* __restrict__ x) {
  int row = blockIdx.x;               // 0..ROWS-1
  int t   = row & (TSEQ - 1);
  int token = idx[row];
  const float* tr = tok + (size_t)token * DMODEL;
  const float* pr = pos + (size_t)t * DMODEL;
  float* xr = x + (size_t)row * DMODEL;
  for (int d = threadIdx.x; d < DMODEL; d += blockDim.x) xr[d] = tr[d] + pr[d];
}

// ---------------------------------------------------------------------------
// Pack per-layer Wq/Wk/Wv [H][D][HS] fp32 -> transposed [H*HS][D] bf16
// (so the GEMM B operand is already [N][K]).
// ---------------------------------------------------------------------------
__global__ void pack_hds_t_kernel(const float* __restrict__ in, uint16_t* __restrict__ out) {
  int o = blockIdx.x * 256 + threadIdx.x;      // o = n*512 + d
  if (o >= DMODEL * DMODEL) return;
  int n = o >> 9;                  // h*64 + s
  int d = o & 511;
  int h = n >> 6, s = n & 63;
  out[o] = f2bf(in[((size_t)h * DMODEL + d) * HSZ + s]);
}

// Tiled transpose+cast: in [K][N] fp32 -> out [N][K] bf16. Grid (N/32, K/32).
__global__ __launch_bounds__(256) void transpose_cast_kernel(
    const float* __restrict__ in, uint16_t* __restrict__ out, int K, int N) {
  __shared__ float tile[32][33];
  const int kb = blockIdx.y * 32, nb = blockIdx.x * 32;
  const int tx = threadIdx.x & 31, ty = threadIdx.x >> 5;   // 32 x 8
#pragma unroll
  for (int i = 0; i < 4; ++i)
    tile[ty + 8 * i][tx] = in[(size_t)(kb + ty + 8 * i) * N + nb + tx];
  __syncthreads();
#pragma unroll
  for (int i = 0; i < 4; ++i)
    out[(size_t)(nb + ty + 8 * i) * K + kb + tx] = f2bf(tile[tx][ty + 8 * i]);
}

// ---------------------------------------------------------------------------
// LayerNorm: one wave32 per row (D=512 -> 16 elems/lane), bf16 output.
// ---------------------------------------------------------------------------
__global__ __launch_bounds__(256) void ln_kernel(const float* __restrict__ x,
                                                 const float* __restrict__ gamma,
                                                 const float* __restrict__ beta,
                                                 uint16_t* __restrict__ out, int rows) {
  int wid = threadIdx.x >> 5, lane = threadIdx.x & 31;
  int row = blockIdx.x * 8 + wid;
  if (row >= rows) return;
  const float* xr = x + (size_t)row * DMODEL;
  float v[16];
  float s = 0.f;
#pragma unroll
  for (int i = 0; i < 16; ++i) { v[i] = xr[lane + 32 * i]; s += v[i]; }
#pragma unroll
  for (int m = 16; m >= 1; m >>= 1) s += __shfl_xor(s, m, 32);
  float mean = s * (1.0f / DMODEL);
  float s2 = 0.f;
#pragma unroll
  for (int i = 0; i < 16; ++i) { float d = v[i] - mean; s2 += d * d; }
#pragma unroll
  for (int m = 16; m >= 1; m >>= 1) s2 += __shfl_xor(s2, m, 32);
  float r = rsqrtf(s2 * (1.0f / DMODEL) + 1e-5f);
  uint16_t* orow = out + (size_t)row * DMODEL;
#pragma unroll
  for (int i = 0; i < 16; ++i) {
    int c = lane + 32 * i;
    orow[c] = f2bf((v[i] - mean) * r * gamma[c] + beta[c]);
  }
}

// ---------------------------------------------------------------------------
// GEMM: C[M,N] = A[M,K](bf16,row) * Bt[N,K](bf16,row; pre-transposed B)
// (+bias)(+resid)(relu?). Block 256 thr = 8 waves; block tile 128x128; wave
// tile 32x64 (2x4 WMMA); K stepped by 32. BOTH tiles staged by async
// global->LDS b128, double-buffered: async issue for tile kt+1 overlaps the
// WMMA compute of tile kt; s_wait_asynccnt 0 + barrier fence the swap.
// Template flags give straight-line fused epilogues.
// Requires M%128==0, N%128==0, K%32==0 (true for all call sites).
// ---------------------------------------------------------------------------
template <int HAS_BIAS, int HAS_RESID, int RELU, int OUTF>
__global__ __launch_bounds__(256) void gemm_bf16_wmma(
    const uint16_t* __restrict__ A, const uint16_t* __restrict__ Bt,
    const float* __restrict__ bias, const float* __restrict__ resid,
    void* __restrict__ out, int M, int N, int K) {
  __shared__ uint16_t As[2][128 * 40];    // [m][k], stride 40 halfs
  __shared__ uint16_t Bts[2][128 * 40];   // [n][k], stride 40 halfs

  const int t    = threadIdx.x;
  const int wid  = t >> 5, lane = t & 31;
  const int grp  = lane >> 4, ln16 = lane & 15;
  const int wm   = wid >> 1, wn = wid & 1;
  const int blockM = blockIdx.y * 128, blockN = blockIdx.x * 128;

  v8f acc[2][4];
#pragma unroll
  for (int i = 0; i < 2; ++i)
#pragma unroll
    for (int j = 0; j < 4; ++j) acc[i][j] = {};

  const int ksteps = K >> 5;

  auto stage = [&](int buf, int kt) {
#pragma unroll
    for (int it = 0; it < 2; ++it) {
      int idx = it * 256 + t;
      int r = idx >> 2, cb = idx & 3;    // 128 rows x 4 chunks of 8 halfs
      async_copy_b128(As[buf] + r * 40 + cb * 8,
                      A + (size_t)(blockM + r) * K + kt * 32 + cb * 8);
      async_copy_b128(Bts[buf] + r * 40 + cb * 8,
                      Bt + (size_t)(blockN + r) * K + kt * 32 + cb * 8);
    }
  };

  stage(0, 0);
  for (int kt = 0; kt < ksteps; ++kt) {
    const int buf = kt & 1;
    wait_async0();        // tile kt landed in LDS (this wave's issues)
    __syncthreads();      // all waves landed; prev buffer free for reuse
    if (kt + 1 < ksteps) stage(buf ^ 1, kt + 1);   // overlap with compute

    Frag fa[2], fb[4];
#pragma unroll
    for (int i = 0; i < 2; ++i)
      load_a(fa[i], As[buf] + (wm * 32 + i * 16 + ln16) * 40, 0, grp);
#pragma unroll
    for (int j = 0; j < 4; ++j)
      load_b_contig(fb[j], Bts[buf] + (wn * 64 + j * 16 + ln16) * 40, 0, grp);
#pragma unroll
    for (int i = 0; i < 2; ++i)
#pragma unroll
      for (int j = 0; j < 4; ++j)
        acc[i][j] = wmma_bf16(fa[i], fb[j], acc[i][j]);
  }

  // Epilogue. C layout: element e -> m = e + 8*(lane/16), n = lane&15.
#pragma unroll
  for (int i = 0; i < 2; ++i) {
#pragma unroll
    for (int j = 0; j < 4; ++j) {
      const int row0 = blockM + wm * 32 + i * 16 + (grp << 3);
      const int col  = blockN + wn * 64 + j * 16 + ln16;
      const float bb = HAS_BIAS ? bias[col] : 0.f;
      float rv[8];
      if (HAS_RESID) {
        const float* rp = resid + (size_t)row0 * N + col;
#pragma unroll
        for (int e = 0; e < 8; ++e) rv[e] = rp[(size_t)e * N];
      }
#pragma unroll
      for (int e = 0; e < 8; ++e) {
        float vv = acc[i][j][e] + bb;
        if (HAS_RESID) vv += rv[e];
        if (RELU) vv = fmaxf(vv, 0.f);
        size_t o = (size_t)(row0 + e) * N + col;
        if (OUTF) ((float*)out)[o] = vv;
        else      ((uint16_t*)out)[o] = f2bf(vv);
      }
    }
  }
}

// ---------------------------------------------------------------------------
// Flash attention (causal). Block = 128 thr (4 waves) handles 64 query rows
// of one (b,h). Each wave owns 16 rows; Q fragments live in registers for
// the whole sweep. K staged by async global->LDS; V staged TRANSPOSED
// (VtT[hs][key]) so P.V B-fragments are contiguous (ds_load_b128). S = Q K^T
// via WMMA; online softmax (row stats via half-wave shfl_xor); P transposed
// through per-wave LDS; O += P V via WMMA. Tiles past the diagonal skipped;
// next tile prefetched (global_prefetch_b8).
// ---------------------------------------------------------------------------
__global__ __launch_bounds__(128) void attn_kernel(const uint16_t* __restrict__ q,
                                                   const uint16_t* __restrict__ k,
                                                   const uint16_t* __restrict__ v,
                                                   uint16_t* __restrict__ o) {
  __shared__ uint16_t Kt[64 * 72];         // [key][hs], stride 72
  __shared__ uint16_t VtT[64 * 72];        // [hs][key], stride 72 (transposed)
  __shared__ uint16_t Pt[4 * 16 * 72];     // per-wave P^bf16 [16 m][64 key]

  const int t    = threadIdx.x;
  const int wid  = t >> 5, lane = t & 31;
  const int grp  = lane >> 4, ln16 = lane & 15;
  const int bh   = blockIdx.y, b = bh / HNUM, h = bh % HNUM;
  const int qbase = blockIdx.x * 64;
  const float scale = 0.125f;              // HS^-0.5

  // Q fragments (rows = lane&15 within this wave's 16-row strip)
  const int qrow_a = qbase + wid * 16 + ln16;
  const uint16_t* qrp = q + ((size_t)(b * TSEQ + qrow_a)) * DMODEL + h * HSZ;
  Frag aq[2];
  load_a(aq[0], qrp, 0, grp);
  load_a(aq[1], qrp, 32, grp);

  float mrun[8], lrun[8];
  v8f accO[4];
#pragma unroll
  for (int e = 0; e < 8; ++e) { mrun[e] = -3.0e38f; lrun[e] = 0.f; }
#pragma unroll
  for (int j = 0; j < 4; ++j) accO[j] = {};

  uint16_t* Ptw = Pt + wid * (16 * 72);

  for (int jt = 0; jt <= (int)blockIdx.x; ++jt) {
    const int jbase = jt * 64;
    __syncthreads();
    // Stage K (async b128) and V transposed (uint4 load + b16 stores)
#pragma unroll
    for (int it = 0; it < 4; ++it) {
      int idx = it * 128 + t;
      int r = idx >> 3, cb = idx & 7;      // 64 rows x 8 chunks of 8 halfs
      size_t src = ((size_t)(b * TSEQ + jbase + r)) * DMODEL + h * HSZ + cb * 8;
      async_copy_b128(Kt + r * 72 + cb * 8, k + src);
      U4 u;
      u.q = *(const uint4*)(v + src);
      if (jt < (int)blockIdx.x) {          // prefetch next tile (64 rows on)
        __builtin_prefetch(k + src + (size_t)64 * DMODEL, 0, 1);
        __builtin_prefetch(v + src + (size_t)64 * DMODEL, 0, 1);
      }
#pragma unroll
      for (int qq = 0; qq < 8; ++qq) VtT[(cb * 8 + qq) * 72 + r] = u.h[qq];
    }
    wait_async0();
    __syncthreads();

    // S = Q K^T : 4 key-tiles x 2 k-steps (HS=64)
    v8f s[4];
#pragma unroll
    for (int nt = 0; nt < 4; ++nt) {
      s[nt] = {};
      Frag fb;
      load_b_contig(fb, Kt + (nt * 16 + ln16) * 72, 0, grp);
      s[nt] = wmma_bf16(aq[0], fb, s[nt]);
      load_b_contig(fb, Kt + (nt * 16 + ln16) * 72, 32, grp);
      s[nt] = wmma_bf16(aq[1], fb, s[nt]);
    }

    // Scale + causal mask + row max
    float rmax[8];
#pragma unroll
    for (int e = 0; e < 8; ++e) rmax[e] = -3.0e38f;
#pragma unroll
    for (int nt = 0; nt < 4; ++nt)
#pragma unroll
      for (int e = 0; e < 8; ++e) {
        int key = jbase + nt * 16 + ln16;
        int qr  = qbase + wid * 16 + (grp << 3) + e;
        float val = (key > qr) ? -3.0e38f : s[nt][e] * scale;
        s[nt][e] = val;
        rmax[e]  = fmaxf(rmax[e], val);
      }
#pragma unroll
    for (int m = 1; m <= 8; m <<= 1)
#pragma unroll
      for (int e = 0; e < 8; ++e) rmax[e] = fmaxf(rmax[e], __shfl_xor(rmax[e], m, 32));

    // Online softmax update
    float alpha[8], rsum[8];
#pragma unroll
    for (int e = 0; e < 8; ++e) {
      float mnew = fmaxf(mrun[e], rmax[e]);
      alpha[e] = __expf(mrun[e] - mnew);
      mrun[e] = mnew;
      rsum[e] = 0.f;
    }
#pragma unroll
    for (int nt = 0; nt < 4; ++nt)
#pragma unroll
      for (int e = 0; e < 8; ++e) {
        float p = __expf(s[nt][e] - mrun[e]);
        rsum[e] += p;
        Ptw[((grp << 3) + e) * 72 + nt * 16 + ln16] = f2bf(p);
      }
#pragma unroll
    for (int m = 1; m <= 8; m <<= 1)
#pragma unroll
      for (int e = 0; e < 8; ++e) rsum[e] += __shfl_xor(rsum[e], m, 32);
#pragma unroll
    for (int e = 0; e < 8; ++e) lrun[e] = lrun[e] * alpha[e] + rsum[e];
#pragma unroll
    for (int nt = 0; nt < 4; ++nt)
#pragma unroll
      for (int e = 0; e < 8; ++e) accO[nt][e] *= alpha[e];

    // O += P V : key contraction (2 k-steps of 32), 4 hs-tiles.
    // Pt is per-wave private; same-wave DS ordering handled via dscnt waits.
#pragma unroll
    for (int p2 = 0; p2 < 2; ++p2) {
      Frag pa;
      load_a(pa, Ptw + ln16 * 72, p2 * 32, grp);
#pragma unroll
      for (int nt = 0; nt < 4; ++nt) {
        Frag vb;
        load_b_contig(vb, VtT + (nt * 16 + ln16) * 72, p2 * 32, grp);
        accO[nt] = wmma_bf16(pa, vb, accO[nt]);
      }
    }
  }

  // Normalize and store bf16 into [B,T,H*HS]
#pragma unroll
  for (int nt = 0; nt < 4; ++nt)
#pragma unroll
    for (int e = 0; e < 8; ++e) {
      int row = qbase + wid * 16 + (grp << 3) + e;
      int col = h * HSZ + nt * 16 + ln16;
      o[((size_t)(b * TSEQ + row)) * DMODEL + col] = f2bf(accO[nt][e] / lrun[e]);
    }
}

// ---------------------------------------------------------------------------
// Host-side launch
// ---------------------------------------------------------------------------
extern "C" void kernel_launch(void* const* d_in, const int* in_sizes, int n_in,
                              void* d_out, int out_size, void* d_ws, size_t ws_size,
                              hipStream_t stream) {
  const int*   idx  = (const int*)d_in[0];
  const float* tok  = (const float*)d_in[1];
  const float* pos  = (const float*)d_in[2];
  const float* Wq   = (const float*)d_in[3];
  const float* Wk   = (const float*)d_in[4];
  const float* Wv   = (const float*)d_in[5];
  const float* Wo   = (const float*)d_in[6];
  const float* bo   = (const float*)d_in[7];
  const float* ln1g = (const float*)d_in[8];
  const float* ln1b = (const float*)d_in[9];
  const float* ln2g = (const float*)d_in[10];
  const float* ln2b = (const float*)d_in[11];
  const float* W1   = (const float*)d_in[12];
  const float* b1   = (const float*)d_in[13];
  const float* W2   = (const float*)d_in[14];
  const float* b2   = (const float*)d_in[15];
  const float* lnfg = (const float*)d_in[16];
  const float* lnfb = (const float*)d_in[17];
  const float* Wlm  = (const float*)d_in[18];
  const float* blm  = (const float*)d_in[19];
  float* out = (float*)d_out;

  char* ws = (char*)d_ws;
  size_t off = 0;
  auto alloc = [&](size_t bytes) -> void* {
    void* p = ws + off;
    off = (off + bytes + 255) & ~(size_t)255;
    return p;
  };
  float*    x    = (float*)   alloc((size_t)ROWS * DMODEL * 4);
  uint16_t* hln  = (uint16_t*)alloc((size_t)ROWS * DMODEL * 2);
  uint16_t* qb   = (uint16_t*)alloc((size_t)ROWS * DMODEL * 2);
  uint16_t* kb   = (uint16_t*)alloc((size_t)ROWS * DMODEL * 2);
  uint16_t* vb   = (uint16_t*)alloc((size_t)ROWS * DMODEL * 2);
  uint16_t* ob   = (uint16_t*)alloc((size_t)ROWS * DMODEL * 2);
  uint16_t* ffh  = (uint16_t*)alloc((size_t)ROWS * DFF * 2);
  uint16_t* wqp  = (uint16_t*)alloc((size_t)DMODEL * DMODEL * 2);   // [N][K]
  uint16_t* wkp  = (uint16_t*)alloc((size_t)DMODEL * DMODEL * 2);
  uint16_t* wvp  = (uint16_t*)alloc((size_t)DMODEL * DMODEL * 2);
  uint16_t* wop  = (uint16_t*)alloc((size_t)DMODEL * DMODEL * 2);
  uint16_t* w1p  = (uint16_t*)alloc((size_t)DMODEL * DFF * 2);
  uint16_t* w2p  = (uint16_t*)alloc((size_t)DFF * DMODEL * 2);
  uint16_t* wlmp = (uint16_t*)alloc((size_t)DMODEL * VOCAB * 2);
  (void)ws_size; (void)in_sizes; (void)n_in; (void)out_size;

  const int DD  = DMODEL * DMODEL;

  embed_kernel<<<ROWS, 256, 0, stream>>>(idx, tok, pos, x);
  // Wlm [512][32000] -> wlmp [32000][512]
  transpose_cast_kernel<<<dim3(VOCAB / 32, DMODEL / 32), 256, 0, stream>>>(Wlm, wlmp,
                                                                           DMODEL, VOCAB);

  dim3 gD(DMODEL / 128, ROWS / 128);
  dim3 gF(DFF / 128, ROWS / 128);
  dim3 gAtt(TSEQ / 64, BATCH * HNUM);

  for (int l = 0; l < LNUM; ++l) {
    pack_hds_t_kernel<<<(DD + 255) / 256, 256, 0, stream>>>(Wq + (size_t)l * HNUM * DMODEL * HSZ, wqp);
    pack_hds_t_kernel<<<(DD + 255) / 256, 256, 0, stream>>>(Wk + (size_t)l * HNUM * DMODEL * HSZ, wkp);
    pack_hds_t_kernel<<<(DD + 255) / 256, 256, 0, stream>>>(Wv + (size_t)l * HNUM * DMODEL * HSZ, wvp);
    transpose_cast_kernel<<<dim3(DMODEL / 32, DMODEL / 32), 256, 0, stream>>>(
        Wo + (size_t)l * DD, wop, DMODEL, DMODEL);
    transpose_cast_kernel<<<dim3(DFF / 32, DMODEL / 32), 256, 0, stream>>>(
        W1 + (size_t)l * DMODEL * DFF, w1p, DMODEL, DFF);
    transpose_cast_kernel<<<dim3(DMODEL / 32, DFF / 32), 256, 0, stream>>>(
        W2 + (size_t)l * DFF * DMODEL, w2p, DFF, DMODEL);

    ln_kernel<<<ROWS / 8, 256, 0, stream>>>(x, ln1g + l * DMODEL, ln1b + l * DMODEL, hln, ROWS);

    // q/k/v = hln @ W  (bf16 out, no bias)
    gemm_bf16_wmma<0, 0, 0, 0><<<gD, 256, 0, stream>>>(hln, wqp, nullptr, nullptr, qb,
                                                       ROWS, DMODEL, DMODEL);
    gemm_bf16_wmma<0, 0, 0, 0><<<gD, 256, 0, stream>>>(hln, wkp, nullptr, nullptr, kb,
                                                       ROWS, DMODEL, DMODEL);
    gemm_bf16_wmma<0, 0, 0, 0><<<gD, 256, 0, stream>>>(hln, wvp, nullptr, nullptr, vb,
                                                       ROWS, DMODEL, DMODEL);

    attn_kernel<<<gAtt, 128, 0, stream>>>(qb, kb, vb, ob);

    // x = x + attn_out @ Wo + bo   (elementwise-safe in-place residual)
    gemm_bf16_wmma<1, 1, 0, 1><<<gD, 256, 0, stream>>>(ob, wop, bo + l * DMODEL, x, x,
                                                       ROWS, DMODEL, DMODEL);

    ln_kernel<<<ROWS / 8, 256, 0, stream>>>(x, ln2g + l * DMODEL, ln2b + l * DMODEL, hln, ROWS);

    // ffh = relu(hln @ W1 + b1)  (bf16 out)
    gemm_bf16_wmma<1, 0, 1, 0><<<gF, 256, 0, stream>>>(hln, w1p, b1 + l * DFF, nullptr, ffh,
                                                       ROWS, DFF, DMODEL);
    // x = x + ffh @ W2 + b2
    gemm_bf16_wmma<1, 1, 0, 1><<<gD, 256, 0, stream>>>(ffh, w2p, b2 + l * DMODEL, x, x,
                                                       ROWS, DMODEL, DFF);
  }

  ln_kernel<<<ROWS / 8, 256, 0, stream>>>(x, lnfg, lnfb, hln, ROWS);

  dim3 gV(VOCAB / 128, ROWS / 128);
  gemm_bf16_wmma<1, 0, 0, 1><<<gV, 256, 0, stream>>>(hln, wlmp, blm, nullptr, out,
                                                     ROWS, VOCAB, DMODEL);
}
